// MultiHeadAttention_11321533792408
// MI455X (gfx1250) — matrix operand order repaired and checked by reading, not compiled, not run on hardware
//
#include <hip/hip_runtime.h>
#include <hip/hip_bf16.h>

#define BATCH 4
#define SEQ   2048
#define EMB   768
#define NH    12
#define HD    64
#define N3    (3*EMB)          // 2304
#define MROWS (BATCH*SEQ)      // 8192

typedef __attribute__((ext_vector_type(16))) __bf16         v16bf;
typedef __attribute__((ext_vector_type(8)))  float          v8f;
typedef __attribute__((ext_vector_type(8)))  unsigned short us8;
typedef __attribute__((ext_vector_type(4)))  int            v4i;
typedef unsigned short u16t;

union FragBF { v16bf bf; us8 h[2]; };
union AccF   { v8f v; float f[8]; };

// CDNA5 async global->LDS copy path (ASYNCcnt-tracked), with safe fallback.
// Probe-derived signature: (AS1 v4i*, AS3 v4i*, imm int offset, imm int cpol).
#if defined(__gfx1250__) && __has_builtin(__builtin_amdgcn_global_load_async_to_lds_b128)
#define ASYNC_CP 1
#define AS1_CAST(p) ((__attribute__((address_space(1))) v4i*)(p))
#define AS3_CAST(p) ((__attribute__((address_space(3))) v4i*)(p))
__device__ __forceinline__ void wait_async0() {
#if __has_builtin(__builtin_amdgcn_s_wait_asynccnt)
  __builtin_amdgcn_s_wait_asynccnt(0);
#else
  asm volatile("s_wait_asynccnt 0x0" ::: "memory");
#endif
}
#endif

__device__ __forceinline__ u16t f2bf(float x) {
  unsigned u = __float_as_uint(x);
  u += 0x7FFFu + ((u >> 16) & 1u);       // round to nearest even
  return (u16t)(u >> 16);
}

// ---------------------------------------------------------------- converts
__global__ __launch_bounds__(256) void cvt_bf16(const float* __restrict__ s,
                                                u16t* __restrict__ d, int n) {
  int i = blockIdx.x * blockDim.x + threadIdx.x;
  int stride = gridDim.x * blockDim.x;
  for (; i < n; i += stride) d[i] = f2bf(s[i]);
}

// d[c*rows + r] = bf16(s[r*cols + c])   (makes weights N-major: B-frag rows contiguous)
__global__ __launch_bounds__(256) void transpose_bf16(const float* __restrict__ s,
                                                      u16t* __restrict__ d,
                                                      int rows, int cols) {
  int i = blockIdx.x * blockDim.x + threadIdx.x;
  int n = rows * cols;
  int stride = gridDim.x * blockDim.x;
  for (; i < n; i += stride) {
    int r = i / cols, c = i - r * cols;
    d[c * rows + r] = f2bf(s[i]);
  }
}

// ---------------------------------------------------------------- QKV GEMM
// Block = 8 waves = 128 M-rows x 64 N-cols. The 64x128 bf16 weight panel is
// staged into LDS with async global->LDS copies and shared by all 8 waves.
__global__ __launch_bounds__(256) void qkv_gemm(const u16t* __restrict__ Xb,
                                                const u16t* __restrict__ WqkvT, // [2304][768]
                                                const float* __restrict__ bqkv,
                                                u16t* __restrict__ Q,
                                                u16t* __restrict__ Kb,
                                                u16t* __restrict__ Vt) {
#if defined(__gfx1250__)
  __shared__ __align__(16) u16t Bpanel[64][128];  // 16 KB
  const int lane = threadIdx.x & 31;
  const int wave = threadIdx.x >> 5;
  const int ln = lane & 15;
  const int hh = lane >> 4;
  const int tilesN4 = N3 / 64;                 // 36
  const int mtg = blockIdx.x / tilesN4;        // 64 groups of 8 M-tiles
  const int n4  = blockIdx.x - mtg * tilesN4;
  const int mt  = mtg * 8 + wave;              // 64*8 = 512 M-tiles exactly

  AccF acc[4];
#pragma unroll
  for (int i = 0; i < 4; i++)
#pragma unroll
    for (int j = 0; j < 8; j++) acc[i].f[j] = 0.f;

  const u16t* aptr  = Xb + (size_t)(mt * 16 + ln) * EMB;
  const u16t* bbase = WqkvT + (size_t)(n4 * 64) * EMB;

  for (int k0 = 0; k0 < EMB; k0 += 128) {
    // ---- stage B panel: 64 rows x 128 K, 1024 x 16B chunks, 4 per thread
#pragma unroll
    for (int c = 0; c < 4; c++) {
      const int chunk = threadIdx.x * 4 + c;
      const int brow = chunk >> 4;
      const int bcol = (chunk & 15) * 8;
      const u16t* gp = bbase + (size_t)brow * EMB + k0 + bcol;
      u16t* lp = &Bpanel[brow][bcol];
#ifdef ASYNC_CP
      __builtin_amdgcn_global_load_async_to_lds_b128(
          AS1_CAST(const_cast<u16t*>(gp)), AS3_CAST(lp), 0, 0);
#else
      *(us8*)lp = *(const us8*)gp;
#endif
    }
#ifdef ASYNC_CP
    wait_async0();
#endif
    __syncthreads();

    // ---- 4 x 32-K WMMA steps off the staged panel
#pragma unroll
    for (int kk = 0; kk < 128; kk += 32) {
      FragBF a;                                // A: lane=M row, 16 contiguous K at 16h
      a.h[0] = *(const us8*)(aptr + k0 + kk + hh * 16);
      a.h[1] = *(const us8*)(aptr + k0 + kk + hh * 16 + 8);
#pragma unroll
      for (int nb = 0; nb < 4; nb++) {
        const u16t* bp = &Bpanel[nb * 16 + ln][kk + hh * 16];
        FragBF bfr;                            // B: lane=N col, 16 contiguous K
        bfr.h[0] = *(const us8*)(bp);
        bfr.h[1] = *(const us8*)(bp + 8);
        acc[nb].v = __builtin_amdgcn_wmma_f32_16x16x32_bf16(
            false, a.bf, false, bfr.bf, (short)0, acc[nb].v, false, false);
      }
    }
    __syncthreads();
  }

#pragma unroll
  for (int nb = 0; nb < 4; nb++) {
    const int n = n4 * 64 + nb * 16 + ln;
    const float bias = bqkv[n];
    const int which = n / EMB;                 // 0=q 1=k 2=v (uniform per 16-tile)
    const int e = n - which * EMB;
    const int head = e >> 6;
    const int dcol = e & 63;
#pragma unroll
    for (int r = 0; r < 8; r++) {
      const int m = mt * 16 + r + 8 * hh;      // C layout: vgpr r -> row r+8h
      const int b = m >> 11;
      const int sp = m & 2047;
      const u16t val = f2bf(acc[nb].f[r] + bias);
      const size_t bh = (size_t)(b * NH + head);
      if (which == 0)      Q [(bh * SEQ + sp) * HD + dcol] = val;
      else if (which == 1) Kb[(bh * SEQ + sp) * HD + dcol] = val;
      else                 Vt[(bh * HD + dcol) * SEQ + sp] = val;
    }
  }
#endif
}

// ---------------------------------------------------------------- flash attention
// one wave = one 16-row query block; online softmax over 32-key steps
__global__ __launch_bounds__(128) void attn_kernel(const u16t* __restrict__ Q,
                                                   const u16t* __restrict__ Kb,
                                                   const u16t* __restrict__ Vt,
                                                   u16t* __restrict__ AOut) {
#if defined(__gfx1250__)
  __shared__ __align__(16) u16t pT[4][16 * 32];   // per-wave P transpose staging
  const int lane = threadIdx.x & 31;
  const int wave = threadIdx.x >> 5;
  const int ln = lane & 15;
  const int hh = lane >> 4;
  const int bh = blockIdx.x >> 5;                 // 48 (b,h) pairs
  const int qg = blockIdx.x & 31;
  const int qb = qg * 4 + wave;                   // 128 q-blocks per (b,h)
  const int b = bh / NH;
  const int h = bh - b * NH;
  const int M0 = qb * 16;

  const u16t* Qbase = Q + ((size_t)bh * SEQ + M0) * HD;
  const u16t* Kbase = Kb + (size_t)bh * SEQ * HD;
  const u16t* Vbase = Vt + (size_t)bh * HD * SEQ;

  FragBF qa0, qa1;                                // Q rows, K(=D) split 0..31 / 32..63
  {
    const u16t* qp = Qbase + ln * HD;
    qa0.h[0] = *(const us8*)(qp + hh * 16);
    qa0.h[1] = *(const us8*)(qp + hh * 16 + 8);
    qa1.h[0] = *(const us8*)(qp + 32 + hh * 16);
    qa1.h[1] = *(const us8*)(qp + 40 + hh * 16);
  }

  AccF oacc[4];
#pragma unroll
  for (int i = 0; i < 4; i++)
#pragma unroll
    for (int j = 0; j < 8; j++) oacc[i].f[j] = 0.f;
  float mrow[8], lrow[8];
#pragma unroll
  for (int r = 0; r < 8; r++) { mrow[r] = -1e30f; lrow[r] = 0.f; }

  const float scale = 0.125f;                     // 1/sqrt(64)

  for (int t0 = 0; t0 < SEQ; t0 += 32) {
    // ---- scores: two 16x16 tiles (keys t0..+15, t0+16..+31), K-dim = D = 2 WMMAs each
    AccF s0, s1;
#pragma unroll
    for (int j = 0; j < 8; j++) { s0.f[j] = 0.f; s1.f[j] = 0.f; }
    {
      const u16t* kpA = Kbase + (size_t)(t0 + ln) * HD + hh * 16;
      const u16t* kpB = kpA + 16 * HD;
      FragBF k00, k01, k10, k11;
      k00.h[0] = *(const us8*)(kpA);      k00.h[1] = *(const us8*)(kpA + 8);
      k01.h[0] = *(const us8*)(kpA + 32); k01.h[1] = *(const us8*)(kpA + 40);
      k10.h[0] = *(const us8*)(kpB);      k10.h[1] = *(const us8*)(kpB + 8);
      k11.h[0] = *(const us8*)(kpB + 32); k11.h[1] = *(const us8*)(kpB + 40);
      s0.v = __builtin_amdgcn_wmma_f32_16x16x32_bf16(false, qa0.bf, false, k00.bf, (short)0, s0.v, false, false);
      s0.v = __builtin_amdgcn_wmma_f32_16x16x32_bf16(false, qa1.bf, false, k01.bf, (short)0, s0.v, false, false);
      s1.v = __builtin_amdgcn_wmma_f32_16x16x32_bf16(false, qa0.bf, false, k10.bf, (short)0, s1.v, false, false);
      s1.v = __builtin_amdgcn_wmma_f32_16x16x32_bf16(false, qa1.bf, false, k11.bf, (short)0, s1.v, false, false);
    }

    // ---- online softmax (rows live across the 16-lane half-groups of the C layout)
#pragma unroll
    for (int r = 0; r < 8; r++) {
      float a0 = s0.f[r] * scale, a1 = s1.f[r] * scale;
      float mx = fmaxf(a0, a1);
#pragma unroll
      for (int off = 1; off < 16; off <<= 1) mx = fmaxf(mx, __shfl_xor(mx, off, 32));
      float mnew = fmaxf(mrow[r], mx);
      float p0 = __expf(a0 - mnew);
      float p1 = __expf(a1 - mnew);
      float sum = p0 + p1;
#pragma unroll
      for (int off = 1; off < 16; off <<= 1) sum += __shfl_xor(sum, off, 32);
      float cor = __expf(mrow[r] - mnew);
      lrow[r] = lrow[r] * cor + sum;
      mrow[r] = mnew;
      s0.f[r] = p0; s1.f[r] = p1;
#pragma unroll
      for (int db = 0; db < 4; db++) oacc[db].f[r] *= cor;
    }

    // ---- transpose P (C layout -> A layout) through LDS
#pragma unroll
    for (int r = 0; r < 8; r++) {
      const int rowp = r + 8 * hh;
      pT[wave][rowp * 32 + ln]      = f2bf(s0.f[r]);
      pT[wave][rowp * 32 + 16 + ln] = f2bf(s1.f[r]);
    }
    __syncthreads();
    FragBF pa;
    pa.h[0] = *(const us8*)(&pT[wave][ln * 32 + hh * 16]);
    pa.h[1] = *(const us8*)(&pT[wave][ln * 32 + hh * 16 + 8]);
    __syncthreads();

    // ---- P(16x32) @ V(32x16) per 16-wide D block; Vt makes lane loads contiguous
#pragma unroll
    for (int db = 0; db < 4; db++) {
      const u16t* vp = Vbase + (size_t)(db * 16 + ln) * SEQ + t0 + hh * 16;
      FragBF vb;
      vb.h[0] = *(const us8*)(vp);
      vb.h[1] = *(const us8*)(vp + 8);
      oacc[db].v = __builtin_amdgcn_wmma_f32_16x16x32_bf16(
          false, pa.bf, false, vb.bf, (short)0, oacc[db].v, false, false);
    }
  }

  // ---- normalize and emit bf16 activations [B*S, E]
#pragma unroll
  for (int r = 0; r < 8; r++) {
    const float inv = 1.0f / lrow[r];
    const size_t grow = (size_t)(b * SEQ + M0 + r + 8 * hh);
#pragma unroll
    for (int db = 0; db < 4; db++)
      AOut[grow * EMB + h * HD + db * 16 + ln] = f2bf(oacc[db].f[r] * inv);
  }
#endif
}

// ---------------------------------------------------------------- output projection
__global__ __launch_bounds__(256) void oproj_gemm(const u16t* __restrict__ Ab,   // [8192][768]
                                                  const u16t* __restrict__ WoT,  // [768][768] N-major
                                                  const float* __restrict__ bo,
                                                  float* __restrict__ out) {
#if defined(__gfx1250__)
  __shared__ __align__(16) u16t Bpanel[64][128];  // 16 KB
  const int lane = threadIdx.x & 31;
  const int wave = threadIdx.x >> 5;
  const int ln = lane & 15;
  const int hh = lane >> 4;
  const int tilesN4 = EMB / 64;                  // 12
  const int mtg = blockIdx.x / tilesN4;          // 64 groups of 8 M-tiles
  const int n4  = blockIdx.x - mtg * tilesN4;
  const int mt  = mtg * 8 + wave;

  AccF acc[4];
#pragma unroll
  for (int i = 0; i < 4; i++)
#pragma unroll
    for (int j = 0; j < 8; j++) acc[i].f[j] = 0.f;

  const u16t* aptr  = Ab + (size_t)(mt * 16 + ln) * EMB;
  const u16t* bbase = WoT + (size_t)(n4 * 64) * EMB;

  for (int k0 = 0; k0 < EMB; k0 += 128) {
#pragma unroll
    for (int c = 0; c < 4; c++) {
      const int chunk = threadIdx.x * 4 + c;
      const int brow = chunk >> 4;
      const int bcol = (chunk & 15) * 8;
      const u16t* gp = bbase + (size_t)brow * EMB + k0 + bcol;
      u16t* lp = &Bpanel[brow][bcol];
#ifdef ASYNC_CP
      __builtin_amdgcn_global_load_async_to_lds_b128(
          AS1_CAST(const_cast<u16t*>(gp)), AS3_CAST(lp), 0, 0);
#else
      *(us8*)lp = *(const us8*)gp;
#endif
    }
#ifdef ASYNC_CP
    wait_async0();
#endif
    __syncthreads();

#pragma unroll
    for (int kk = 0; kk < 128; kk += 32) {
      FragBF a;
      a.h[0] = *(const us8*)(aptr + k0 + kk + hh * 16);
      a.h[1] = *(const us8*)(aptr + k0 + kk + hh * 16 + 8);
#pragma unroll
      for (int nb = 0; nb < 4; nb++) {
        const u16t* bp = &Bpanel[nb * 16 + ln][kk + hh * 16];
        FragBF bfr;
        bfr.h[0] = *(const us8*)(bp);
        bfr.h[1] = *(const us8*)(bp + 8);
        acc[nb].v = __builtin_amdgcn_wmma_f32_16x16x32_bf16(
            false, a.bf, false, bfr.bf, (short)0, acc[nb].v, false, false);
      }
    }
    __syncthreads();
  }

#pragma unroll
  for (int nb = 0; nb < 4; nb++) {
    const int n = n4 * 64 + nb * 16 + ln;
    const float bias = bo[n];
#pragma unroll
    for (int r = 0; r < 8; r++) {
      const int m = mt * 16 + r + 8 * hh;
      out[(size_t)m * EMB + n] = acc[nb].f[r] + bias;
    }
  }
#endif
}

// ---------------------------------------------------------------- launcher
extern "C" void kernel_launch(void* const* d_in, const int* in_sizes, int n_in,
                              void* d_out, int out_size, void* d_ws, size_t ws_size,
                              hipStream_t stream) {
  const float* x    = (const float*)d_in[0];
  const float* Wqkv = (const float*)d_in[1];
  const float* bqkv = (const float*)d_in[2];
  const float* Wo   = (const float*)d_in[3];
  const float* bo   = (const float*)d_in[4];
  float* out = (float*)d_out;

  char* ws = (char*)d_ws;
  // workspace layout (bytes), all 256B aligned; total ~55 MB
  u16t* Xb    = (u16t*)(ws + 0);          // x bf16 [8192][768]; reused as AOut later
  u16t* WqkvT = (u16t*)(ws + 12582912);   // [2304][768] bf16
  u16t* WoT   = (u16t*)(ws + 16121856);   // [768][768] bf16
  u16t* Qb    = (u16t*)(ws + 17301504);   // [B,H,S,D] bf16
  u16t* Kbf   = (u16t*)(ws + 29884416);   // [B,H,S,D] bf16
  u16t* Vt    = (u16t*)(ws + 42467328);   // [B,H,D,S] bf16

  cvt_bf16<<<4096, 256, 0, stream>>>(x, Xb, MROWS * EMB);
  transpose_bf16<<<2048, 256, 0, stream>>>(Wqkv, WqkvT, EMB, N3);
  transpose_bf16<<<1024, 256, 0, stream>>>(Wo, WoT, EMB, EMB);
  qkv_gemm<<<2304, 256, 0, stream>>>(Xb, WqkvT, bqkv, Qb, Kbf, Vt);
  attn_kernel<<<1536, 128, 0, stream>>>(Qb, Kbf, Vt, Xb /* AOut: reuse Xb region */);
  oproj_gemm<<<768, 256, 0, stream>>>(Xb, WoT, bo, out);
}
